// SelfAttention_44315472560877
// MI455X (gfx1250) — compile-verified, matrix operand order checked by reading, not run on hardware
//
#include <hip/hip_runtime.h>
#include <hip/hip_bf16.h>
#include <math.h>

typedef __bf16 bf16_t;
typedef __attribute__((ext_vector_type(8)))  __bf16 v8bf;
typedef __attribute__((ext_vector_type(16))) __bf16 v16bf;
typedef __attribute__((ext_vector_type(8)))  float  v8f;
typedef unsigned int       u32;
typedef unsigned long long u64;
typedef __attribute__((ext_vector_type(4))) u32 u32x4;
typedef __attribute__((ext_vector_type(8))) u32 u32x8;

constexpr int S_LEN = 4096;
constexpr int DM    = 512;
constexpr int NH    = 8;
constexpr int DK    = 64;
constexpr int BATCH = 2;
constexpr int MROWS = BATCH * S_LEN;           // 8192 flattened (b,s) rows

// ---------------------------------------------------------------------------
// WMMA helper: D = A(16x32 bf16) * B(32x16 bf16) + C(16x16 f32)
// ---------------------------------------------------------------------------
__device__ __forceinline__ v8f wmma_bf16(v16bf a, v16bf b, v8f c) {
  return __builtin_amdgcn_wmma_f32_16x16x32_bf16(
      /*neg_a=*/false, a, /*neg_b=*/false, b,
      /*c_mod=*/(short)0, c, /*reuse_a=*/false, /*reuse_b=*/false);
}

// Load a 16x32 bf16 A-fragment (or symmetric B-fragment) from row-major
// [row, k] storage with leading dimension `ld` (elements).
// ISA layout: lane L holds row L%16; halves 0..7 = k in [o, o+8),
// halves 8..15 = k in [o+16, o+24), o = 8*(L>=16).  -> two b128 loads.
__device__ __forceinline__ v16bf ld_frag(const bf16_t* base, int ld, int row,
                                         int k0, int lane) {
  const bf16_t* p = base + (size_t)(row + (lane & 15)) * ld
                         + (k0 + ((lane >> 4) << 3));
  v8bf lo = *(const v8bf*)p;
  v8bf hi = *(const v8bf*)(p + 16);
  return __builtin_shufflevector(lo, hi, 0,1,2,3,4,5,6,7,8,9,10,11,12,13,14,15);
}

// ---------------------------------------------------------------------------
// f32 -> bf16 conversion (grid-stride)
// ---------------------------------------------------------------------------
__global__ void cvt_bf16_kernel(const float* __restrict__ in,
                                bf16_t* __restrict__ out, int n) {
  for (int i = blockIdx.x * blockDim.x + threadIdx.x; i < n;
       i += gridDim.x * blockDim.x)
    out[i] = (bf16_t)in[i];
}

// ---------------------------------------------------------------------------
// QKV projection: Y[m,n] = X[m,:] . W[n,:] + bias[n], scaled, scattered to
// [B, H, S, DK] bf16.  Block = 8 waves (2x4), 32x128 tile; each wave does two
// 16x16 tiles sharing one A fragment per K-step.
// ---------------------------------------------------------------------------
__global__ __launch_bounds__(256)
void proj_qkv_kernel(const bf16_t* __restrict__ X, const bf16_t* __restrict__ W,
                     const float* __restrict__ bias, bf16_t* __restrict__ out,
                     float scale) {
  const int wave = threadIdx.x >> 5, lane = threadIdx.x & 31;
  const int row0 = blockIdx.x * 32 + (wave >> 2) * 16;
  const int col0 = blockIdx.y * 128 + (wave & 3) * 32;
  v8f acc0 = {}, acc1 = {};
#pragma unroll 4
  for (int k0 = 0; k0 < DM; k0 += 32) {
    __builtin_prefetch(X + (size_t)(row0 + (lane & 15)) * DM + k0 + 64, 0, 0);
    v16bf a  = ld_frag(X, DM, row0, k0, lane);
    v16bf b0 = ld_frag(W, DM, col0, k0, lane);
    v16bf b1 = ld_frag(W, DM, col0 + 16, k0, lane);
    acc0 = wmma_bf16(a, b0, acc0);
    acc1 = wmma_bf16(a, b1, acc1);
  }
  const int n16 = lane & 15, hf = lane >> 4;
#pragma unroll
  for (int t = 0; t < 2; ++t) {
    const int n = col0 + 16 * t + n16;
    const int h = n >> 6, d = n & (DK - 1);
    const float bv = bias[n];
    const v8f& acc = t ? acc1 : acc0;
#pragma unroll
    for (int r = 0; r < 8; ++r) {
      const int m = row0 + r + 8 * hf;          // flattened (b, s)
      const int bb = m >> 12, s = m & (S_LEN - 1);
      out[(((size_t)bb * NH + h) * S_LEN + s) * DK + d] =
          (bf16_t)((acc[r] + bv) * scale);
    }
  }
}

// ---------------------------------------------------------------------------
// Output projection: f32 epilogue straight into d_out.
// ---------------------------------------------------------------------------
__global__ __launch_bounds__(256)
void proj_out_kernel(const bf16_t* __restrict__ X, const bf16_t* __restrict__ W,
                     const float* __restrict__ bias, float* __restrict__ out) {
  const int wave = threadIdx.x >> 5, lane = threadIdx.x & 31;
  const int row0 = blockIdx.x * 32 + (wave >> 2) * 16;
  const int col0 = blockIdx.y * 128 + (wave & 3) * 32;
  v8f acc0 = {}, acc1 = {};
#pragma unroll 4
  for (int k0 = 0; k0 < DM; k0 += 32) {
    v16bf a  = ld_frag(X, DM, row0, k0, lane);
    v16bf b0 = ld_frag(W, DM, col0, k0, lane);
    v16bf b1 = ld_frag(W, DM, col0 + 16, k0, lane);
    acc0 = wmma_bf16(a, b0, acc0);
    acc1 = wmma_bf16(a, b1, acc1);
  }
  const int n16 = lane & 15, hf = lane >> 4;
#pragma unroll
  for (int t = 0; t < 2; ++t) {
    const int n = col0 + 16 * t + n16;
    const float bv = bias[n];
    const v8f& acc = t ? acc1 : acc0;
#pragma unroll
    for (int r = 0; r < 8; ++r) {
      const int m = row0 + r + 8 * hf;
      out[(size_t)m * DM + n] = acc[r] + bv;
    }
  }
}

// ---------------------------------------------------------------------------
// Flash attention: grid (B*H, S/128).  8 waves x 16 queries each, streaming
// 32-key blocks with DOUBLE-BUFFERED LDS tiles: the Tensor Data Mover pulls
// K(t+1) and the lanes stage V(t+1) (transposed) while tile t is computed;
// one barrier + s_wait_tensorcnt per iteration.  Q was pre-scaled by
// log2(e)/sqrt(DK) so softmax uses bare v_exp (exp2).  Row sums of P come
// from a WMMA against a ones-matrix.  Output: [B, S, H*DK] bf16.
// ---------------------------------------------------------------------------
__global__ __launch_bounds__(256)
void attn_kernel(const bf16_t* __restrict__ Q, const bf16_t* __restrict__ K,
                 const bf16_t* __restrict__ V, bf16_t* __restrict__ O) {
  __shared__ bf16_t Kt[2][32 * DK];    // [buf][key][dk]   8 KB (TDM target)
  __shared__ bf16_t Vt[2][DK * 32];    // [buf][dk][key]   8 KB
  __shared__ bf16_t Ps[8][16 * 32];    // per-wave P       8 KB

  const int bh = blockIdx.x;
  const int b = bh >> 3, h = bh & 7;
  const size_t hoff = (size_t)bh * S_LEN * DK;
  const bf16_t* Qh = Q + hoff;
  const bf16_t* Kh = K + hoff;
  const bf16_t* Vh = V + hoff;

  const int wave = threadIdx.x >> 5, lane = threadIdx.x & 31;
  const int n16 = lane & 15, hf = lane >> 4;
  const int q0 = blockIdx.y * 128 + wave * 16;

  // Loop-invariant part of the Tensor DMA descriptor (CDNA5 ISA):
  // 2D tensor, 2-byte elements, 64-wide rows, 64x32 tile.
  u32x8 g1;
  g1[0] = 0x00010000u;                          // data_size = 2 bytes
  g1[1] = (u32)DK << 16;                        // tensor_dim0 = 64
  g1[2] = (u32)S_LEN << 16;                     // tensor_dim1 = 4096
  g1[3] = (u32)DK << 16;                        // tile_dim0 = 64
  g1[4] = 32u;                                  // tile_dim1 = 32
  g1[5] = (u32)DK;                              // tensor_dim0_stride = 64
  g1[6] = 0u;
  g1[7] = 0u;
  const u32x4 gz = {0u, 0u, 0u, 0u};            // groups 2/3: 2D tensor

  // Issue async K-tile DMA into buffer `buf` (wave 0 only; EXEC-independent).
  auto issue_k = [&](int key0, int buf) {
    if (wave == 0) {
      const u64 ga = (u64)(uintptr_t)(Kh + (size_t)key0 * DK);
      u32x4 g0;
      g0[0] = 1u;                                   // count=1, user descriptor
      g0[1] = (u32)(uintptr_t)(&Kt[buf][0]);        // lds_addr (addr[31:0])
      g0[2] = (u32)ga;                              // global_addr[31:0]
      g0[3] = (u32)(ga >> 32) | 0x80000000u;        // addr[56:32] | type=2
      asm volatile("tensor_load_to_lds %0, %1, %2, %3"
                   :: "s"(g0), "s"(g1), "s"(gz), "s"(gz)
                   : "memory");
    }
  };
  // Stage V tile transposed into Vt[buf]: Vt[d][key], all 256 lanes.
  auto stage_v = [&](int key0, int buf) {
    const int d = threadIdx.x & 63, k8 = (threadIdx.x >> 6) * 8;
    v8bf vv;
#pragma unroll
    for (int i = 0; i < 8; ++i)
      vv[i] = Vh[(size_t)(key0 + k8 + i) * DK + d];
    *(v8bf*)&Vt[buf][d * 32 + k8] = vv;
  };

  // Per-wave Q fragments (16 rows x 64 dk), resident for the whole kernel.
  v16bf qf0 = ld_frag(Qh, DK, q0, 0, lane);
  v16bf qf1 = ld_frag(Qh, DK, q0, 32, lane);

  // ones B-fragment for row-sum-by-WMMA
  v16bf ones;
#pragma unroll
  for (int i = 0; i < 16; ++i) ones[i] = (bf16_t)1.0f;

  v8f o0 = {}, o1 = {}, o2 = {}, o3 = {};
  float mrow[8], lrow[8], alph[8];
#pragma unroll
  for (int r = 0; r < 8; ++r) { mrow[r] = -INFINITY; lrow[r] = 0.0f; }

  // Prologue: fill buffer 0.
  issue_k(0, 0);
  stage_v(0, 0);
  if (wave == 0) __builtin_amdgcn_s_wait_tensorcnt(0);
  __syncthreads();

  for (int key0 = 0; key0 < S_LEN; key0 += 32) {
    const int buf = (key0 >> 5) & 1;
    const int knext = key0 + 32;
    const bool more = knext < S_LEN;

    // Kick off next tile's DMA + V staging; both hide under this tile's math.
    if (more) { issue_k(knext, buf ^ 1); stage_v(knext, buf ^ 1); }

    const bf16_t* Kb = &Kt[buf][0];
    const bf16_t* Vb = &Vt[buf][0];

    // scores: 16 queries x 32 keys (two 16x16 tiles), dk reduced via 2 WMMAs
    v8f s0 = {}, s1 = {};
    s0 = wmma_bf16(qf0, ld_frag(Kb, DK, 0, 0, lane), s0);
    s0 = wmma_bf16(qf1, ld_frag(Kb, DK, 0, 32, lane), s0);
    s1 = wmma_bf16(qf0, ld_frag(Kb, DK, 16, 0, lane), s1);
    s1 = wmma_bf16(qf1, ld_frag(Kb, DK, 16, 32, lane), s1);

    // online softmax in log2 domain (Q pre-scaled by log2e/sqrt(dk));
    // C-layout: reg r / half hf -> row m = r + 8*hf, lane n = key.
#pragma unroll
    for (int r = 0; r < 8; ++r) {
      float a0 = s0[r], a1 = s1[r];
      float mx = fmaxf(a0, a1);
      mx = fmaxf(mx, __shfl_xor(mx, 1, 16));
      mx = fmaxf(mx, __shfl_xor(mx, 2, 16));
      mx = fmaxf(mx, __shfl_xor(mx, 4, 16));
      mx = fmaxf(mx, __shfl_xor(mx, 8, 16));
      const float mnew = fmaxf(mrow[r], mx);
      alph[r] = exp2f(mrow[r] - mnew);
      mrow[r] = mnew;
      const float p0 = exp2f(a0 - mnew), p1 = exp2f(a1 - mnew);
      const int m = r + 8 * hf;
      Ps[wave][m * 32 + n16]      = (bf16_t)p0;
      Ps[wave][m * 32 + 16 + n16] = (bf16_t)p1;
    }
    // per-wave LDS: DS ops are in-order within a wave; just stop reordering
    asm volatile("" ::: "memory");

    v16bf pf = ld_frag(&Ps[wave][0], 32, 0, 0, lane);

    // row sums via WMMA against ones: every lane of row m holds sum_k P[m,k]
    v8f rsum = {};
    rsum = wmma_bf16(pf, ones, rsum);
#pragma unroll
    for (int r = 0; r < 8; ++r) {
      lrow[r] = lrow[r] * alph[r] + rsum[r];
      o0[r] *= alph[r]; o1[r] *= alph[r]; o2[r] *= alph[r]; o3[r] *= alph[r];
    }

    // P (16x32) x V (32x64): 4 WMMAs over the d dimension
    o0 = wmma_bf16(pf, ld_frag(Vb, 32, 0,  0, lane), o0);
    o1 = wmma_bf16(pf, ld_frag(Vb, 32, 16, 0, lane), o1);
    o2 = wmma_bf16(pf, ld_frag(Vb, 32, 32, 0, lane), o2);
    o3 = wmma_bf16(pf, ld_frag(Vb, 32, 48, 0, lane), o3);

    // Next tile must be fully landed before anyone reads buf^1.
    if (more && wave == 0) __builtin_amdgcn_s_wait_tensorcnt(0);
    __syncthreads();
  }

  // normalize and write [b, s, h*64 + d]
#pragma unroll
  for (int r = 0; r < 8; ++r) {
    const float inv = 1.0f / lrow[r];
    const int m = r + 8 * hf;
    const size_t rowO = ((size_t)b * S_LEN + q0 + m) * DM + h * DK;
    O[rowO + 0  + n16] = (bf16_t)(o0[r] * inv);
    O[rowO + 16 + n16] = (bf16_t)(o1[r] * inv);
    O[rowO + 32 + n16] = (bf16_t)(o2[r] * inv);
    O[rowO + 48 + n16] = (bf16_t)(o3[r] * inv);
  }
}

// ---------------------------------------------------------------------------
// Host launcher.  Inputs: x, Wq, bq, Wk, bk, Wv, bv, Wo, bo (all f32).
// Workspace layout (bf16 elements):
//   [0, NX)                : xbf, later reused as attention output
//   [NX, NX+4*NW)          : Wq, Wk, Wv, Wo in bf16
//   [NX+4*NW, NX+4*NW+3*NX): Q, K, V in [B,H,S,DK]
// Total ~34 MB.
// ---------------------------------------------------------------------------
extern "C" void kernel_launch(void* const* d_in, const int* in_sizes, int n_in,
                              void* d_out, int out_size, void* d_ws,
                              size_t ws_size, hipStream_t stream) {
  (void)in_sizes; (void)n_in; (void)out_size; (void)ws_size;
  const float* x  = (const float*)d_in[0];
  const float* Wq = (const float*)d_in[1];
  const float* bq = (const float*)d_in[2];
  const float* Wk = (const float*)d_in[3];
  const float* bk = (const float*)d_in[4];
  const float* Wv = (const float*)d_in[5];
  const float* bv = (const float*)d_in[6];
  const float* Wo = (const float*)d_in[7];
  const float* bo = (const float*)d_in[8];
  float* out = (float*)d_out;

  constexpr size_t NX = (size_t)MROWS * DM;   // 4,194,304
  constexpr size_t NW = (size_t)DM * DM;      //   262,144

  bf16_t* wsb  = (bf16_t*)d_ws;
  bf16_t* xbf  = wsb;             // also attention output buffer (reused)
  bf16_t* wqb  = wsb + NX;
  bf16_t* wkb  = wqb + NW;
  bf16_t* wvb  = wkb + NW;
  bf16_t* wob  = wvb + NW;
  bf16_t* Qw   = wob + NW;
  bf16_t* Kw   = Qw + NX;
  bf16_t* Vw   = Kw + NX;
  bf16_t* attn = xbf;

  cvt_bf16_kernel<<<1024, 256, 0, stream>>>(x,  xbf, (int)NX);
  cvt_bf16_kernel<<<256,  256, 0, stream>>>(Wq, wqb, (int)NW);
  cvt_bf16_kernel<<<256,  256, 0, stream>>>(Wk, wkb, (int)NW);
  cvt_bf16_kernel<<<256,  256, 0, stream>>>(Wv, wvb, (int)NW);
  cvt_bf16_kernel<<<256,  256, 0, stream>>>(Wo, wob, (int)NW);

  // Q carries both the 1/sqrt(dk) softmax scale and log2(e) for exp2-domain
  // online softmax: 0.125 * 1.4426950408889634.
  dim3 pgrid(MROWS / 32, DM / 128);   // 256 x 4
  proj_qkv_kernel<<<pgrid, 256, 0, stream>>>(xbf, wqb, bq, Qw,
                                             0.125f * 1.4426950408889634f);
  proj_qkv_kernel<<<pgrid, 256, 0, stream>>>(xbf, wkb, bk, Kw, 1.0f);
  proj_qkv_kernel<<<pgrid, 256, 0, stream>>>(xbf, wvb, bv, Vw, 1.0f);

  dim3 agrid(BATCH * NH, S_LEN / 128);  // 16 x 32
  attn_kernel<<<agrid, 256, 0, stream>>>(Qw, Kw, Vw, attn);

  proj_out_kernel<<<pgrid, 256, 0, stream>>>(attn, wob, bo, out);
}
